// TaskAlignedAssigner_34780645163338
// MI455X (gfx1250) — compile-verified, matrix operand order checked by reading, not run on hardware
//
#include <hip/hip_runtime.h>
#include <hip/hip_bf16.h>

// ---------------- problem constants (from reference) ----------------
#define BS_   32
#define NA_   8400          // 80^2 + 40^2 + 20^2 anchors
#define NM_   64            // max GT per image
#define NC_   80            // classes
#define TOPK_ 13

// ---------------- d_out layout (flat float, tuple order) ------------
// (target_labels, target_bboxes, target_scores, fg_mask, target_gt_idx)
#define OUT_L 0
#define OUT_B (BS_*NA_)                       // 268800
#define OUT_S (OUT_B + BS_*NA_*4)             // 1344000
#define OUT_F (OUT_S + BS_*NA_*NC_)           // 22848000
#define OUT_G (OUT_F + BS_*NA_)               // 23116800

// ---------------- workspace layout (bytes) --------------------------
#define NBMA      (BS_*NM_*NA_)               // 17,203,200
#define OFF_ALIGN ((size_t)0)
#define OFF_OVL   (OFF_ALIGN + (size_t)NBMA*4)
#define OFF_MP    (OFF_OVL   + (size_t)NBMA*4)      // u8 mask_pos
#define OFF_FG    (OFF_MP    + (size_t)NBMA)
#define OFF_PA    (OFF_FG    + (size_t)BS_*NA_*4)   // pos_align (bs*nm)
#define OFF_PO    (OFF_PA    + (size_t)BS_*NM_*4)
#define OFF_NORM  (OFF_PO    + (size_t)BS_*NM_*4)

typedef __attribute__((ext_vector_type(16))) _Float16 v16h;
typedef __attribute__((ext_vector_type(8)))  float    v8f;
typedef __attribute__((ext_vector_type(4)))  unsigned v4u;
typedef __attribute__((ext_vector_type(8)))  int      v8i;
typedef __attribute__((ext_vector_type(4)))  int      v4i;

// ---- Tensor Data Mover availability / arity detection --------------
#if defined(__has_builtin)
#if __has_builtin(__builtin_amdgcn_tensor_load_to_lds)
#define HAVE_TDM_ 1
#endif
#endif
#if defined(__has_include)
#if __has_include(<hip/amd_detail/amd_gfx1250_TDM.h>)
#define TDM_6ARG_ 1      // therock-10.0 headers => clang-23 6-arg builtin
#endif
#endif

// --------------------------------------------------------------------
__device__ __forceinline__ float ciou_f(float g0, float g1, float g2, float g3,
                                        float p0, float p1, float p2, float p3) {
  const float eps = 1e-7f;
  float w1 = g2 - g0, h1 = g3 - g1 + eps;
  float w2 = p2 - p0, h2 = p3 - p1 + eps;
  float iw = fmaxf(fminf(g2, p2) - fmaxf(g0, p0), 0.f);
  float ih = fmaxf(fminf(g3, p3) - fmaxf(g1, p1), 0.f);
  float inter = iw * ih;
  float uni = w1 * h1 + w2 * h2 - inter + eps;
  float iou = inter / uni;
  float cw = fmaxf(g2, p2) - fminf(g0, p0);
  float ch = fmaxf(g3, p3) - fminf(g1, p1);
  float c2 = cw * cw + ch * ch + eps;
  float dx = p0 + p2 - g0 - g2;
  float dy = p1 + p3 - g1 - g3;
  float rho2 = (dx * dx + dy * dy) * 0.25f;
  float dat = atanf(w2 / h2) - atanf(w1 / h1);
  float v = 0.40528473456935108578f * dat * dat;        // 4/pi^2
  float aa = v / (v - iou + 1.0f + eps);                // stop_gradient irrelevant fwd
  return iou - (rho2 / c2 + v * aa);
}

// K1: align_metric & masked overlaps, one thread per (b,m,a)
__global__ void k_metrics(const float* __restrict__ ps, const float* __restrict__ pb,
                          const float* __restrict__ anc, const int* __restrict__ glab,
                          const float* __restrict__ gbox, const float* __restrict__ mgt,
                          float* __restrict__ alignm, float* __restrict__ ovl) {
  int a = blockIdx.x * 256 + threadIdx.x;
  int m = blockIdx.y, b = blockIdx.z;
  if (a >= NA_) return;
  int bm = b * NM_ + m;
  int oi = bm * NA_ + a;
  float mg = mgt[bm];
  float g0 = gbox[bm * 4 + 0], g1 = gbox[bm * 4 + 1];
  float g2 = gbox[bm * 4 + 2], g3 = gbox[bm * 4 + 3];
  float ax = anc[2 * a], ay = anc[2 * a + 1];
  float dmin = fminf(fminf(ax - g0, ay - g1), fminf(g2 - ax, g3 - ay));
  bool act = (mg > 0.f) && (dmin > 1e-9f);
  float al = 0.f, ov = 0.f;
  if (act) {
    const float* pbp = pb + (size_t)(b * NA_ + a) * 4;
    __builtin_prefetch(pbp, 0, 0);                      // global_prefetch_b8
    float p0 = pbp[0], p1 = pbp[1], p2 = pbp[2], p3 = pbp[3];
    int lbl = glab[bm];
    float sc = ps[(size_t)(b * NA_ + a) * NC_ + lbl];
    float o = fmaxf(ciou_f(g0, g1, g2, g3, p0, p1, p2, p3), 0.f);
    float q = o * o;
    ov = o;
    al = sc * q * q * q;                                // score^1 * iou^6
  }
  alignm[oi] = al;
  ovl[oi] = ov;
}

// K2: top-13 per (b,m) row with jax-style tie-break (smallest index).
// Row staged into LDS by the Tensor Data Mover (1D tile, TENSORcnt-tracked);
// 13 rounds of block argmax; winners -> mask_pos.
__global__ void k_topk(const float* __restrict__ alignm, const float* __restrict__ anc,
                       const float* __restrict__ gbox, const float* __restrict__ mgt,
                       unsigned char* __restrict__ mp) {
  int m = blockIdx.x, b = blockIdx.y;
  int bm = b * NM_ + m;
  if (mgt[bm] <= 0.f) return;                            // block-uniform exit
  __shared__ float lv[33 * 256];                         // lv[a] = align row value
  __shared__ float rv[256];
  __shared__ int   ri[256];
  int t = threadIdx.x;
  const float* row = alignm + (size_t)bm * NA_;

#if defined(HAVE_TDM_)
  if (t < 32) {                                          // wave 0 issues the DMA
    unsigned lds_off = (unsigned)(__SIZE_TYPE__)(__attribute__((address_space(3))) void*)&lv[0];
    unsigned long long ga = (unsigned long long)(__SIZE_TYPE__)row;
    // D# group 0: count=1 | lds_addr | global_addr[56:0] | type=2
    v4u g0;
    g0[0] = 1u;
    g0[1] = lds_off;
    g0[2] = (unsigned)(ga & 0xffffffffu);
    g0[3] = (unsigned)((ga >> 32) & 0x01ffffffu) | (2u << 30);
    // D# group 1: data_size=4B; tensor_dim0=8400; tensor_dim1=1;
    //             tile_dim0=8400 (1D tile); tensor_dim0_stride=8400
    v8i g1;
    g1[0] = 0x00020000;                  // [17:16] data_size=2 (4 bytes)
    g1[1] = (int)(0x20D0u << 16);        // tensor_dim0[15:0]=8400 in bits [63:48]->word1[31:16]
    g1[2] = 0x00010000;                  // tensor_dim0 hi=0 | tensor_dim1 lo16 = 1
    g1[3] = (int)(0x20D0u << 16);        // tensor_dim1 hi=0 | tile_dim0 = 8400
    g1[4] = 0;                           // tile_dim1=0 (unused), tile_dim2=0
    g1[5] = 8400;                        // tensor_dim0_stride low32
    g1[6] = 0;
    g1[7] = 0;
    v4i g2 = {0, 0, 0, 0};               // <=2D tensor: groups 2/3 unused
    v4i g3 = {0, 0, 0, 0};
#if defined(TDM_6ARG_)
    v8i g4 = {0, 0, 0, 0, 0, 0, 0, 0};
    __builtin_amdgcn_tensor_load_to_lds(g0, g1, g2, g3, g4, 0);
#else
    __builtin_amdgcn_tensor_load_to_lds(g0, g1, g2, g3, 0);
#endif
    __builtin_amdgcn_s_wait_tensorcnt(0);                // issuing wave waits
  }
  // OOB tail sentinels (a in [8400, 8448))
  {
    int a = 32 * 256 + t;
    if (a >= NA_) lv[a] = -1.0f;
  }
#else
  for (int j = 0; j < 33; ++j) {
    int a = j * 256 + t;
    lv[a] = (a < NA_) ? row[a] : -1.0f;
  }
#endif

  float g0b = gbox[bm * 4 + 0], g1b = gbox[bm * 4 + 1];
  float g2b = gbox[bm * 4 + 2], g3b = gbox[bm * 4 + 3];
  __syncthreads();                                       // TDM data visible to all waves

  for (int k = 0; k < TOPK_; ++k) {
    float bv = -3.f; int bi = 0;
    for (int j = 0; j < 33; ++j) {
      float v = lv[j * 256 + t];
      if (v > bv) { bv = v; bi = j * 256 + t; }          // ascending a: first-max kept
    }
    rv[t] = bv; ri[t] = bi;
    __syncthreads();
    for (int s = 128; s > 0; s >>= 1) {
      if (t < s) {
        if (rv[t + s] > rv[t] || (rv[t + s] == rv[t] && ri[t + s] < ri[t])) {
          rv[t] = rv[t + s]; ri[t] = ri[t + s];
        }
      }
      __syncthreads();
    }
    int wi = ri[0];
    if (t == (wi & 255)) lv[wi] = -2.f;                  // exclude winner
    if (t == 0) {
      float ax = anc[2 * wi], ay = anc[2 * wi + 1];
      float dmin = fminf(fminf(ax - g0b, ay - g1b), fminf(g2b - ax, g3b - ay));
      mp[(size_t)bm * NA_ + wi] = (dmin > 1e-9f) ? 1 : 0; // mask_topk * mask_in_gts
    }
    __syncthreads();
  }
}

// K3: fg[b,a] = sum_m mask_pos[b,m,a] via WMMA f16 (ones(16x32) x mask(32x16)).
// mask values {0,1} are exact in f16; sums <= 64 exact in f32 accumulators.
// One wave per 16-anchor tile; NA_ = 525 * 16 exactly (no tail lanes).
__global__ void k_fg_wmma(const unsigned char* __restrict__ mp, float* __restrict__ fg) {
  int b = blockIdx.y;
  int wave = threadIdx.x >> 5;
  int lane = threadIdx.x & 31;
  int tile = blockIdx.x * 8 + wave;
  if (tile >= NA_ / 16) return;                          // wave-uniform exit -> EXEC all-1s
  int a0 = tile * 16;
  int n = lane & 15;                                     // B column (anchor within tile)
  int kh = (lane >> 4) << 4;                             // lanes 0-15: K 0-15; 16-31: K 16-31
  v16h A;
#pragma unroll
  for (int e = 0; e < 16; ++e) A[e] = (_Float16)1.0f;
  v8f c = {};
#pragma unroll
  for (int kb = 0; kb < NM_; kb += 32) {
    v16h Bm;
#pragma unroll
    for (int e = 0; e < 16; ++e) {
      int k = kb + kh + e;
      Bm[e] = (_Float16)(float)mp[(size_t)(b * NM_ + k) * NA_ + (a0 + n)];
    }
    c = __builtin_amdgcn_wmma_f32_16x16x32_f16(false, A, false, Bm,
                                               (short)0, c, false, false);
  }
  // D layout: VGPR0, lanes 0-15 = row M=0, N=lane (all rows identical here)
  if (lane < 16) fg[(size_t)b * NA_ + a0 + lane] = c[0];
}

// K4: per-anchor conflict resolution; rewrites mask_pos; emits labels/bboxes/fg/tgi.
__global__ void k_resolve(const float* __restrict__ ovl, unsigned char* __restrict__ mp,
                          const float* __restrict__ fg, const int* __restrict__ glab,
                          const float* __restrict__ gbox, float* __restrict__ out) {
  int id = blockIdx.x * 256 + threadIdx.x;
  if (id >= BS_ * NA_) return;
  int b = id / NA_, a = id - b * NA_;
  float bestov = -1e30f; int mo = 0; int firstpos = -1;
  for (int m = 0; m < NM_; ++m) {
    int i = (b * NM_ + m) * NA_ + a;
    float o = ovl[i];
    if (o > bestov) { bestov = o; mo = m; }              // first max (jnp.argmax)
    if (firstpos < 0 && mp[i]) firstpos = m;
  }
  float fgv = fg[id];
  int tgi; float fgm;
  if (fgv > 1.f) {                                       // multi-assigned anchor
    tgi = mo; fgm = 1.f;
    for (int m = 0; m < NM_; ++m)
      mp[(b * NM_ + m) * NA_ + a] = (m == mo) ? 1 : 0;   // one_hot(argmax overlaps)
  } else {
    tgi = (firstpos >= 0) ? firstpos : 0;
    fgm = (firstpos >= 0) ? 1.f : 0.f;
  }
  int bm = b * NM_ + tgi;
  int lbl = glab[bm]; if (lbl < 0) lbl = 0;
  out[OUT_L + id] = (float)lbl;
  out[OUT_B + (size_t)id * 4 + 0] = gbox[bm * 4 + 0];
  out[OUT_B + (size_t)id * 4 + 1] = gbox[bm * 4 + 1];
  out[OUT_B + (size_t)id * 4 + 2] = gbox[bm * 4 + 2];
  out[OUT_B + (size_t)id * 4 + 3] = gbox[bm * 4 + 3];
  out[OUT_F + id] = fgm;
  out[OUT_G + id] = (float)tgi;
}

// K5: pos_align / pos_ovl = max over anchors of (align|ovl)*mask_pos per (b,m)
__global__ void k_rowred(const float* __restrict__ alignm, const float* __restrict__ ovl,
                         const unsigned char* __restrict__ mp,
                         float* __restrict__ pa, float* __restrict__ po) {
  int m = blockIdx.x, b = blockIdx.y;
  int bm = b * NM_ + m;
  int t = threadIdx.x;
  __shared__ float sa[256], so[256];
  float ma = 0.f, mo = 0.f;                              // masked values are >= 0
  for (int a = t; a < NA_; a += 256) {
    int i = bm * NA_ + a;
    if (mp[i]) { ma = fmaxf(ma, alignm[i]); mo = fmaxf(mo, ovl[i]); }
  }
  sa[t] = ma; so[t] = mo;
  __syncthreads();
  for (int s = 128; s > 0; s >>= 1) {
    if (t < s) { sa[t] = fmaxf(sa[t], sa[t + s]); so[t] = fmaxf(so[t], so[t + s]); }
    __syncthreads();
  }
  if (t == 0) { pa[bm] = sa[0]; po[bm] = so[0]; }
}

// K6a: per-anchor normalizer
__global__ void k_norm(const float* __restrict__ alignm, const unsigned char* __restrict__ mp,
                       const float* __restrict__ pa, const float* __restrict__ po,
                       float* __restrict__ normv) {
  int id = blockIdx.x * 256 + threadIdx.x;
  if (id >= BS_ * NA_) return;
  int b = id / NA_, a = id - b * NA_;
  float nv = 0.f;
  for (int m = 0; m < NM_; ++m) {
    int bm = b * NM_ + m;
    int i = bm * NA_ + a;
    if (mp[i]) {
      float cand = alignm[i] * po[bm] / (pa[bm] + 1e-9f);
      nv = fmaxf(nv, cand);
    }
  }
  normv[id] = nv;
}

// K6b: target_scores = one_hot(label) * norm, gated by fg_mask (coalesced 86MB store)
__global__ void k_scores(const float* __restrict__ normv, float* __restrict__ out) {
  int idx = blockIdx.x * 256 + threadIdx.x;
  if (idx >= BS_ * NA_ * NC_) return;
  int c = idx % NC_;
  int ba = idx / NC_;
  int lbl = (int)out[OUT_L + ba];
  float fgv = out[OUT_F + ba];
  out[OUT_S + (size_t)idx] = (fgv > 0.5f && c == lbl) ? normv[ba] : 0.f;
}

// --------------------------------------------------------------------
extern "C" void kernel_launch(void* const* d_in, const int* in_sizes, int n_in,
                              void* d_out, int out_size, void* d_ws, size_t ws_size,
                              hipStream_t stream) {
  const float* ps   = (const float*)d_in[0];   // pd_scores (bs,na,C)
  const float* pb   = (const float*)d_in[1];   // pd_bboxes (bs,na,4)
  const float* anc  = (const float*)d_in[2];   // anc_points (na,2)
  const int*   glab = (const int*)d_in[3];     // gt_labels (bs,nm,1) int32
  // d_in[4] = gt_scores: unused by reference
  const float* gbox = (const float*)d_in[5];   // gt_bboxes (bs,nm,4)
  const float* mgt  = (const float*)d_in[6];   // mask_gt (bs,nm,1)
  float* out = (float*)d_out;

  char* ws = (char*)d_ws;
  float*         w_align = (float*)(ws + OFF_ALIGN);
  float*         w_ovl   = (float*)(ws + OFF_OVL);
  unsigned char* w_mp    = (unsigned char*)(ws + OFF_MP);
  float*         w_fg    = (float*)(ws + OFF_FG);
  float*         w_pa    = (float*)(ws + OFF_PA);
  float*         w_po    = (float*)(ws + OFF_PO);
  float*         w_norm  = (float*)(ws + OFF_NORM);

  // zero mask_pos (graph-capturable async memset)
  hipMemsetAsync(w_mp, 0, (size_t)NBMA, stream);

  k_metrics<<<dim3((NA_ + 255) / 256, NM_, BS_), 256, 0, stream>>>(
      ps, pb, anc, glab, gbox, mgt, w_align, w_ovl);

  k_topk<<<dim3(NM_, BS_), 256, 0, stream>>>(w_align, anc, gbox, mgt, w_mp);

  k_fg_wmma<<<dim3((NA_ / 16 + 7) / 8, BS_), 256, 0, stream>>>(w_mp, w_fg);

  k_resolve<<<dim3((BS_ * NA_ + 255) / 256), 256, 0, stream>>>(
      w_ovl, w_mp, w_fg, glab, gbox, out);

  k_rowred<<<dim3(NM_, BS_), 256, 0, stream>>>(w_align, w_ovl, w_mp, w_pa, w_po);

  k_norm<<<dim3((BS_ * NA_ + 255) / 256), 256, 0, stream>>>(
      w_align, w_mp, w_pa, w_po, w_norm);

  k_scores<<<dim3((BS_ * NA_ * NC_ + 255) / 256), 256, 0, stream>>>(w_norm, out);
}
